// TopKSoftmaxFaiss_80564996539146
// MI455X (gfx1250) — matrix operand order repaired and checked by reading, not compiled, not run on hardware
//
#include <hip/hip_runtime.h>
#include <stdint.h>

typedef __attribute__((ext_vector_type(8)))  float  v8f;
typedef __attribute__((ext_vector_type(16))) __bf16 v16bf;

#define TOKENS 4096
#define DMODEL 1024
#define VOCAB  32000
#define KSEL   128

#define TT 32              // tokens per workgroup (two WMMA M-tiles)
#define CAP 256            // candidate pool slots per token (power of 2 for bitonic)
#define COMPACT_AT 192     // compact when count exceeds this (headroom = 64)
#define XPAD 8             // LDS row pad (bf16 elems) -> 16B-aligned rows, conflict spread
#define VITER 256          // vocab columns per WG iteration (8 waves x 2 N-tiles x 16)
#define VHALF (VOCAB / 2)  // 16000 vocab rows per WG (split across gridDim.y)
#define ITERS_FULL (VHALF / VITER)          // 62 full iterations
#define VTAIL (VHALF - ITERS_FULL * VITER)  // 128 tail rows (single N-tile pass)

union Frag { uint4 u[2]; v16bf v; };

__device__ __forceinline__ unsigned short f2bf(float f) {
    unsigned int u = __float_as_uint(f);
    u += 0x7fffu + ((u >> 16) & 1u);     // round-to-nearest-even
    return (unsigned short)(u >> 16);
}

#define WMMA_BF16(A, B, C) \
    __builtin_amdgcn_wmma_f32_16x16x32_bf16(false, (A), false, (B), (short)0, (C), false, false)

// ---- prep: convert W (f32) -> bf16 in workspace, fused ||w||^2 row norms ----
__global__ __launch_bounds__(256) void knn_prep(const float* __restrict__ w,
                                                unsigned short* __restrict__ bbf,
                                                float* __restrict__ bsq) {
    const int lane = threadIdx.x & 31;
    const int wave = threadIdx.x >> 5;
    const int row  = blockIdx.x * 8 + wave;          // 4000 blocks * 8 rows = 32000
    const float* src = w + (size_t)row * DMODEL;
    unsigned short* dst = bbf + (size_t)row * DMODEL;
    float s = 0.f;
#pragma unroll
    for (int i = 0; i < 8; ++i) {
        const int idx = i * 128 + lane * 4;          // coalesced float4
        float4 v = *(const float4*)(src + idx);
        s += v.x * v.x + v.y * v.y + v.z * v.z + v.w * v.w;
        uint2 p;
        p.x = (unsigned)f2bf(v.x) | ((unsigned)f2bf(v.y) << 16);
        p.y = (unsigned)f2bf(v.z) | ((unsigned)f2bf(v.w) << 16);
        *(uint2*)(dst + idx) = p;
    }
#pragma unroll
    for (int off = 16; off > 0; off >>= 1) s += __shfl_xor(s, off, 32);
    if (lane == 0) bsq[row] = s;
}

// ---- workgroup-wide bitonic sort of all TT token pools (ascending) ----
__device__ __forceinline__ void sort_pools(float (*pool)[CAP], int tid) {
    const int tpt = 256 / TT;                 // threads per token (8)
    const int ppt = (CAP / 2) / tpt;          // pairs per thread (16)
    const int tt  = tid / tpt;
    const int sub = tid % tpt;
    for (int k = 2; k <= CAP; k <<= 1) {
        for (int j = k >> 1; j > 0; j >>= 1) {
#pragma unroll
            for (int p = 0; p < ppt; ++p) {
                int pid = sub * ppt + p;
                int i   = ((pid & ~(j - 1)) << 1) | (pid & (j - 1));
                int ip  = i | j;
                bool up = ((i & k) == 0);
                float a = pool[tt][i], b = pool[tt][ip];
                if ((a > b) == up) { pool[tt][i] = b; pool[tt][ip] = a; }
            }
            __syncthreads();
        }
    }
}

// ---- main: fused bf16-WMMA distance GEMM (2Mx2N register tile) + streaming top-k ----
__global__ __launch_bounds__(256, 1) void knn_main(const float* __restrict__ x,
                                                   const unsigned short* __restrict__ bbf,
                                                   const float* __restrict__ bsq,
                                                   float* __restrict__ part) {
    __shared__ unsigned short xlds[TT][DMODEL + XPAD];  // 66 KB: x tile as bf16
    __shared__ float pool[TT][CAP];                     // 32 KB: candidate pools
    __shared__ float xsq[TT];
    __shared__ int   cnt[TT];
    __shared__ float thr[TT];
    __shared__ int   flag;

    const int tid  = threadIdx.x;
    const int lane = tid & 31;
    const int wave = tid >> 5;
    const int n    = lane & 15;
    const int hi   = lane >> 4;
    const int tok0 = blockIdx.x * TT;
    const int half = blockIdx.y;
    const int vhb  = half * VHALF;

    if (tid < TT) { xsq[tid] = 0.f; cnt[tid] = 0; thr[tid] = __builtin_inff(); }
    for (int i = tid; i < TT * CAP; i += 256) ((float*)pool)[i] = __builtin_inff();
    __syncthreads();

    // stage x tile -> LDS bf16, accumulate ||x||^2
    {
        const int t  = tid >> 3;                  // 8 threads per token
        const int c0 = (tid & 7) * 128;
        const float* xr = x + (size_t)(tok0 + t) * DMODEL + c0;
        float s = 0.f;
#pragma unroll
        for (int i = 0; i < 32; ++i) {
            float4 v = *(const float4*)(xr + i * 4);
            s += v.x * v.x + v.y * v.y + v.z * v.z + v.w * v.w;
            xlds[t][c0 + i * 4 + 0] = f2bf(v.x);
            xlds[t][c0 + i * 4 + 1] = f2bf(v.y);
            xlds[t][c0 + i * 4 + 2] = f2bf(v.z);
            xlds[t][c0 + i * 4 + 3] = f2bf(v.w);
        }
        atomicAdd(&xsq[t], s);
    }
    __syncthreads();

    auto maybe_compact = [&]() {
        if (tid == 0) {
            int mx = 0;
            for (int t = 0; t < TT; ++t) mx = (cnt[t] > mx) ? cnt[t] : mx;
            flag = (mx > COMPACT_AT);
        }
        __syncthreads();
        if (flag) {
            sort_pools(pool, tid);   // ends with a barrier
            if (tid < TT) {
                int c = cnt[tid] < KSEL ? cnt[tid] : KSEL;
                cnt[tid] = c;
                thr[tid] = pool[tid][KSEL - 1];
            }
            for (int i = tid; i < TT * (CAP - KSEL); i += 256) {
                int t = i / (CAP - KSEL);
                int j = KSEL + i % (CAP - KSEL);
                pool[t][j] = __builtin_inff();
            }
            __syncthreads();
        }
    };

    // distances carry only (-2*dot + ||w||^2); ||x||^2 is a per-token constant
    // (ranking-invariant), added back at emit time.
    auto ins1 = [&](float d, int t) {
        if (d < thr[t]) {
            int pos = atomicAdd(&cnt[t], 1);      // ds_add_rtn_u32
            if (pos < CAP) pool[t][pos] = d;
        }
    };

    const int koffA = hi * 8;   // 16-bit A layout: lanes 0-15 K{0..7,16..23}, 16-31 K{8..15,24..31}
    const unsigned short* a0p = &xlds[n][koffA];         // M-tile 0 (tokens 0..15)
    const unsigned short* a1p = &xlds[16 + n][koffA];    // M-tile 1 (tokens 16..31)

    for (int iter = 0; iter < ITERS_FULL; ++iter) {
        const int vb = vhb + iter * VITER + wave * 32;
        const unsigned short* brow0 = bbf + (size_t)(vb + n) * DMODEL + hi * 8;
        const unsigned short* brow1 = brow0 + (size_t)16 * DMODEL;
        __builtin_prefetch(bbf + (size_t)(vb + VITER + n) * DMODEL, 0, 1);  // global_prefetch_b8

        v8f c00 = {}, c01 = {}, c10 = {}, c11 = {};
        Frag A0[2], A1[2], B0[2], B1[2];
        auto ldf = [&](int kk, int buf) {
            const int k0 = kk * 32;
            A0[buf].u[0] = *(const uint4*)(a0p + k0);        // ds_load_b128
            A0[buf].u[1] = *(const uint4*)(a0p + k0 + 16);
            A1[buf].u[0] = *(const uint4*)(a1p + k0);
            A1[buf].u[1] = *(const uint4*)(a1p + k0 + 16);
            B0[buf].u[0] = *(const uint4*)(brow0 + k0);      // global_load_b128 (L2 resident)
            B0[buf].u[1] = *(const uint4*)(brow0 + k0 + 16);
            B1[buf].u[0] = *(const uint4*)(brow1 + k0);
            B1[buf].u[1] = *(const uint4*)(brow1 + k0 + 16);
        };
        ldf(0, 0);
        // pipelined 2Mx2N: each A frag feeds 2 WMMAs, each B frag feeds 2 WMMAs
#pragma unroll 4
        for (int kk = 0; kk < DMODEL / 32; ++kk) {
            const int cur = kk & 1, nxt = cur ^ 1;
            if (kk + 1 < DMODEL / 32) ldf(kk + 1, nxt);
            c00 = WMMA_BF16(A0[cur].v, B0[cur].v, c00);
            c10 = WMMA_BF16(A1[cur].v, B0[cur].v, c10);
            c01 = WMMA_BF16(A0[cur].v, B1[cur].v, c01);
            c11 = WMMA_BF16(A1[cur].v, B1[cur].v, c11);
        }

        const float vs0 = bsq[vb + n];
        const float vs1 = bsq[vb + 16 + n];
        float d00[8], d01[8], d10[8], d11[8];
#pragma unroll
        for (int r = 0; r < 8; ++r) {
            d00[r] = fmaf(-2.0f, c00[r], vs0);
            d01[r] = fmaf(-2.0f, c01[r], vs1);
            d10[r] = fmaf(-2.0f, c10[r], vs0);
            d11[r] = fmaf(-2.0f, c11[r], vs1);
        }
        // 4 phases of 2 waves: <=64 candidates/token/phase -> never overflows CAP
#pragma unroll
        for (int p = 0; p < 4; ++p) {
            if ((wave >> 1) == p) {
#pragma unroll
                for (int r = 0; r < 8; ++r) {
                    ins1(d00[r], r + hi * 8);
                    ins1(d01[r], r + hi * 8);
                    ins1(d10[r], 16 + r + hi * 8);
                    ins1(d11[r], 16 + r + hi * 8);
                }
            }
            __syncthreads();
            maybe_compact();
        }
    }

    // tail: last VTAIL=128 vocab rows, single N-tile per wave
    {
        const int vb = vhb + ITERS_FULL * VITER + wave * 16;
        const unsigned short* brow = bbf + (size_t)(vb + n) * DMODEL + hi * 8;
        v8f c0 = {}, c1 = {};
        Frag A0[2], A1[2], B[2];
        auto ldf = [&](int kk, int buf) {
            const int k0 = kk * 32;
            A0[buf].u[0] = *(const uint4*)(a0p + k0);
            A0[buf].u[1] = *(const uint4*)(a0p + k0 + 16);
            A1[buf].u[0] = *(const uint4*)(a1p + k0);
            A1[buf].u[1] = *(const uint4*)(a1p + k0 + 16);
            B[buf].u[0]  = *(const uint4*)(brow + k0);
            B[buf].u[1]  = *(const uint4*)(brow + k0 + 16);
        };
        ldf(0, 0);
#pragma unroll 4
        for (int kk = 0; kk < DMODEL / 32; ++kk) {
            const int cur = kk & 1, nxt = cur ^ 1;
            if (kk + 1 < DMODEL / 32) ldf(kk + 1, nxt);
            c0 = WMMA_BF16(A0[cur].v, B[cur].v, c0);
            c1 = WMMA_BF16(A1[cur].v, B[cur].v, c1);
        }
        const float vs = bsq[vb + n];
        float d0[8], d1[8];
#pragma unroll
        for (int r = 0; r < 8; ++r) {
            d0[r] = fmaf(-2.0f, c0[r], vs);
            d1[r] = fmaf(-2.0f, c1[r], vs);
        }
#pragma unroll
        for (int p = 0; p < 2; ++p) {
            if ((wave >> 2) == p) {
#pragma unroll
                for (int r = 0; r < 8; ++r) {
                    ins1(d0[r], r + hi * 8);
                    ins1(d1[r], 16 + r + hi * 8);
                }
            }
            __syncthreads();
            maybe_compact();
        }
    }

    // final canonical sort; emit sorted top-128 (+||x||^2) per token to workspace
    sort_pools(pool, tid);
    for (int i = tid; i < TT * KSEL; i += 256) {
        int t = i >> 7, j = i & (KSEL - 1);
        part[((size_t)(tok0 + t) * 2 + half) * KSEL + j] = pool[t][j] + xsq[t];
    }
}

// ---- merge two sorted 128-lists per token (ascending + reversed = bitonic) ----
__global__ __launch_bounds__(128) void knn_merge(const float* __restrict__ part,
                                                 float* __restrict__ out) {
    __shared__ float buf[2 * KSEL];
    const int t = blockIdx.x;
    const int tid = threadIdx.x;
    buf[tid]        = part[(size_t)t * 2 * KSEL + tid];
    buf[KSEL + tid] = part[(size_t)t * 2 * KSEL + KSEL + (KSEL - 1 - tid)];
    __syncthreads();
    for (int j = KSEL; j > 0; j >>= 1) {          // bitonic merge, 8 substages
        int i  = ((tid & ~(j - 1)) << 1) | (tid & (j - 1));
        int ip = i | j;
        float a = buf[i], b = buf[ip];
        if (a > b) { buf[i] = b; buf[ip] = a; }
        __syncthreads();
    }
    out[(size_t)t * KSEL + tid] = buf[tid];
}

extern "C" void kernel_launch(void* const* d_in, const int* in_sizes, int n_in,
                              void* d_out, int out_size, void* d_ws, size_t ws_size,
                              hipStream_t stream) {
    const float* x = (const float*)d_in[0];
    const float* w = (const float*)d_in[1];
    char* ws = (char*)d_ws;
    unsigned short* bbf = (unsigned short*)ws;                        // 64 MB bf16 W
    float* bsq  = (float*)(ws + (size_t)VOCAB * DMODEL * 2);          // 128 KB row norms
    float* part = (float*)(ws + (size_t)VOCAB * DMODEL * 2 + VOCAB * 4); // 4 MB partial top-k

    knn_prep<<<VOCAB / 8, 256, 0, stream>>>(w, bbf, bsq);
    dim3 grid(TOKENS / TT, 2);
    knn_main<<<grid, 256, 0, stream>>>(x, bbf, bsq, part);
    knn_merge<<<TOKENS, 128, 0, stream>>>(part, (float*)d_out);
}
